// Embeddings1DConnector_26027501814133
// MI455X (gfx1250) — compile-verified, matrix operand order checked by reading, not run on hardware
//
#include <hip/hip_runtime.h>
#include <hip/hip_bf16.h>

// ---------------------------------------------------------------------------
// MI455X (gfx1250) implementation. Memory-bound on fp32 weight streaming
// (~1.42 GB/call -> ~61us @ 23.3 TB/s); all matmuls use
// v_wmma_f32_16x16x32_bf16 (wave32, 16x16 tiles, f32 accumulate).
// GEMM A-tiles are streamed by the Tensor Data Mover (tensor_load_to_lds,
// TENSORcnt-fenced) into double-buffered LDS; W-tiles are fp32->bf16
// converted in-register on the way into LDS.
// ---------------------------------------------------------------------------

#define Bc   2
#define Sc   1024
#define Dc   3840
#define Hc   30
#define DHc  128
#define DFFc 15360
#define Rc   128
#define BDIM 256

typedef __attribute__((ext_vector_type(16))) __bf16 v16bf;
typedef __attribute__((ext_vector_type(8)))  float  v8f;
typedef __attribute__((ext_vector_type(8)))  unsigned short us8;
typedef __attribute__((ext_vector_type(4)))  float  f4;
typedef __attribute__((ext_vector_type(4)))  unsigned int u32x4;
typedef __attribute__((ext_vector_type(8)))  int i32x8;
typedef __attribute__((ext_vector_type(4)))  int i32x4;

#if defined(__has_builtin)
#if __has_builtin(__builtin_amdgcn_tensor_load_to_lds)
#define HAVE_TDM 1
#endif
#endif
#ifndef HAVE_TDM
#define HAVE_TDM 0
#endif

__device__ __forceinline__ unsigned short f2bf(float f) {
  union { float f; unsigned u; } v; v.f = f;
  unsigned r = (v.u + 0x7FFFu + ((v.u >> 16) & 1u)) >> 16;  // RNE
  return (unsigned short)r;
}
__device__ __forceinline__ float bf2f(unsigned short s) {
  union { unsigned u; float f; } v; v.u = ((unsigned)s) << 16;
  return v.f;
}
// Build a 16x32 (A) / 32x16 (B) bf16 fragment: each lane holds two
// 8-element contiguous K runs (ISA 7.12.2 16-bit layout).
__device__ __forceinline__ v16bf ld_frag(const unsigned short* p0,
                                         const unsigned short* p1) {
  union { v16bf v; us8 h[2]; } r;
  r.h[0] = *(const us8*)p0;
  r.h[1] = *(const us8*)p1;
  return r.v;
}

#if HAVE_TDM
// TDM: DMA a [rows=64] x [cols=32] bf16 tile (row stride = K elements) from
// global into LDS (rows stored back-to-back, 32 elems each). D# bit layout
// per CDNA5 ISA 8.3/8.4: 2-D tensor, data_size=2B, tile_dim0=32, tile_dim1=64.
__device__ __forceinline__ void tdm_load_a(const unsigned short* gptr,
                                           unsigned lds_off, int K) {
  unsigned long long ga = (unsigned long long)(size_t)gptr;
  u32x4 g0;
  g0[0] = 1u;                                    // count=1, user descriptor
  g0[1] = lds_off;                               // lds_addr [63:32]
  g0[2] = (unsigned)(ga & 0xFFFFFFFFu);          // global_addr [95:64]
  g0[3] = (unsigned)((ga >> 32) & 0x1FFFFFFu)    // global_addr [120:96]
          | (2u << 30);                          // type=2 ("image") [127:126]
  i32x8 g1;
  g1[0] = (1 << 16);                 // wg_mask=0, data_size=1 (2 bytes)
  g1[1] = (32 << 16);                // tensor_dim0 lo16 @ [63:48]
  g1[2] = (64 << 16);                // tensor_dim0 hi | tensor_dim1 lo16
  g1[3] = (32 << 16);                // tensor_dim1 hi | tile_dim0=32
  g1[4] = 64;                        // tile_dim1=64, tile_dim2=0
  g1[5] = K;                         // tensor_dim0_stride lo32
  g1[6] = 0;                         // stride0 hi16 | stride1 lo16
  g1[7] = 0;                         // stride1 hi32
  i32x4 z = {0, 0, 0, 0};
#if __clang_major__ >= 23
  i32x8 z8 = {0, 0, 0, 0, 0, 0, 0, 0};
  __builtin_amdgcn_tensor_load_to_lds(g0, g1, z, z, z8, 0);
#else
  __builtin_amdgcn_tensor_load_to_lds(g0, g1, z, z, 0);
#endif
}
#endif

enum { EPI_BF16 = 0, EPI_GELU = 1, EPI_RESID = 2 };

// ---------------------------------------------------------------------------
// GEMM: out[M,N] = A[M,K](bf16) x W[K,N](fp32, converted to bf16 on the fly)
// Block: 256 thr = 8 waves. Tile BM=64, BN=128, BK=32, double-buffered LDS.
// Wave w owns N-stripe w*16 and computes 4 M-subtiles (B-frag reuse x4).
// A tile via TDM (wave 0 issues, s_wait_tensorcnt fences); W tile via
// cooperative load + f32->bf16 + transposed ds_store_b32.
// ---------------------------------------------------------------------------
template <int EPI>
__global__ __launch_bounds__(BDIM) void gemm_wmma(
    const unsigned short* __restrict__ A,
    const float* __restrict__ W,
    unsigned short* __restrict__ outB,
    float* __restrict__ outF,
    const float* __restrict__ resid,
    int M, int N, int K)
{
  __shared__ unsigned short lds_a[2][64][32];    // A tile, row-major [m][k]
  __shared__ unsigned short lds_bt[2][128][32];  // W tile transposed [n][k]
  const int tid   = threadIdx.x;
  const int wave  = tid >> 5, lane = tid & 31;
  const int lhalf = lane >> 4, lmod = lane & 15, kbase = lhalf * 8;
  const int m0 = blockIdx.y * 64, n0 = blockIdx.x * 128;
  const int nk = K >> 5;

  // Stage fill: A tile (TDM or cooperative), W tile (convert + transpose).
  auto load_a = [&](int kt, int stage) {
#if HAVE_TDM
    if (wave == 0)
      tdm_load_a(&A[(size_t)m0 * K + kt * 32],
                 (unsigned)(size_t)&lds_a[stage][0][0], K);
#else
    int r = tid >> 2, c = (tid & 3) * 8;
    *(us8*)&lds_a[stage][r][c] =
        *(const us8*)&A[(size_t)(m0 + r) * K + kt * 32 + c];
#endif
  };
  auto load_w = [&](int kt, int stage) {
    int kk = (tid >> 4) * 2, nc = (tid & 15) * 8;   // 2 k-rows x 8 n each
    const float* s0v = &W[(size_t)(kt * 32 + kk) * N + n0 + nc];
    const float* s1v = s0v + N;
    f4 a0 = *(const f4*)(s0v);
    f4 a1 = *(const f4*)(s0v + 4);
    f4 b0 = *(const f4*)(s1v);
    f4 b1 = *(const f4*)(s1v + 4);
    #pragma unroll
    for (int i = 0; i < 4; ++i) {
      unsigned p0 = (unsigned)f2bf(a0[i]) | ((unsigned)f2bf(b0[i]) << 16);
      unsigned p1 = (unsigned)f2bf(a1[i]) | ((unsigned)f2bf(b1[i]) << 16);
      *(unsigned*)&lds_bt[stage][nc + i][kk]     = p0;  // bf16 pair (k,k+1)
      *(unsigned*)&lds_bt[stage][nc + 4 + i][kk] = p1;
    }
  };
  auto fence_a = [&]() {
#if HAVE_TDM
    if (wave == 0) __builtin_amdgcn_s_wait_tensorcnt(0);
#endif
  };

  v8f acc[4] = {};
  load_a(0, 0);
  load_w(0, 0);
  fence_a();
  __syncthreads();

  for (int kt = 0; kt < nk; ++kt) {
    const int cur = kt & 1;
    v16bf bfrag = ld_frag(&lds_bt[cur][wave * 16 + lmod][kbase],
                          &lds_bt[cur][wave * 16 + lmod][kbase + 16]);
    #pragma unroll
    for (int mi = 0; mi < 4; ++mi) {
      v16bf afrag = ld_frag(&lds_a[cur][mi * 16 + lmod][kbase],
                            &lds_a[cur][mi * 16 + lmod][kbase + 16]);
      acc[mi] = __builtin_amdgcn_wmma_f32_16x16x32_bf16(
          false, afrag, false, bfrag, (short)0, acc[mi], false, false);
    }
    if (kt + 1 < nk) {
      load_a(kt + 1, cur ^ 1);
      load_w(kt + 1, cur ^ 1);
      if (kt + 2 < nk)
        __builtin_prefetch(&W[(size_t)((kt + 2) * 32) * N + n0], 0, 1);
      fence_a();
    }
    __syncthreads();
  }
  // Epilogue: C layout -> m = r + 8*(lane/16), n = lane%16 (ISA 7.12.2)
  #pragma unroll
  for (int mi = 0; mi < 4; ++mi) {
    #pragma unroll
    for (int r = 0; r < 8; ++r) {
      int m = m0 + mi * 16 + lhalf * 8 + r;
      int n = n0 + wave * 16 + lmod;
      size_t idx = (size_t)m * N + n;
      float v = acc[mi][r];
      if constexpr (EPI == EPI_RESID) {
        outF[idx] = resid[idx] + v;
      } else if constexpr (EPI == EPI_GELU) {
        float g = 0.5f * v *
            (1.0f + tanhf(0.7978845608028654f * (v + 0.044715f * v * v * v)));
        outB[idx] = f2bf(g);
      } else {
        outB[idx] = f2bf(v);
      }
    }
  }
}

// ---------------------------------------------------------------------------
// Flash attention (mask is all-zero after register replacement).
// Block = 8 waves = 128 query rows of one (b,h). Key loop in tiles of 32.
// QK^T and PV both via WMMA; P converted C->A layout through per-wave LDS.
// ---------------------------------------------------------------------------
__global__ __launch_bounds__(BDIM) void attn_kernel(
    const unsigned short* __restrict__ q,
    const unsigned short* __restrict__ k,
    const unsigned short* __restrict__ v,
    unsigned short* __restrict__ o_out)
{
  __shared__ unsigned short lds_k[32][128];    // [key][dh]
  __shared__ unsigned short lds_vt[128][32];   // [dh][key]
  __shared__ unsigned short lds_p[8][16][32];  // per wave: [qrow][key]
  const int tid   = threadIdx.x;
  const int wave  = tid >> 5, lane = tid & 31;
  const int lhalf = lane >> 4, lmod = lane & 15, kbase = lhalf * 8;
  const int b = blockIdx.x / Hc, h = blockIdx.x % Hc;
  const int q0 = blockIdx.y * 128 + wave * 16;

  v16bf qf[4];
  {
    const unsigned short* qp =
        q + (size_t)(b * Sc + q0 + lmod) * Dc + h * DHc;
    #pragma unroll
    for (int kc = 0; kc < 4; ++kc)
      qf[kc] = ld_frag(qp + kc * 32 + kbase, qp + kc * 32 + kbase + 16);
  }
  v8f o[8] = {};
  float mrow[8], lrow[8];
  #pragma unroll
  for (int r = 0; r < 8; ++r) { mrow[r] = -3.0e38f; lrow[r] = 0.0f; }
  const float scale = 0.08838834764831845f;  // 1/sqrt(128)

  for (int kt = 0; kt < Sc / 32; ++kt) {
    __syncthreads();
    #pragma unroll
    for (int it = 0; it < 2; ++it) {  // 512 16B chunks, 2/thread
      int c = tid * 2 + it;
      int row = c >> 4, dc = (c & 15) * 8;
      size_t g = (size_t)(b * Sc + kt * 32 + row) * Dc + h * DHc + dc;
      *(us8*)&lds_k[row][dc] = *(const us8*)&k[g];
      us8 vv = *(const us8*)&v[g];
      #pragma unroll
      for (int e = 0; e < 8; ++e) lds_vt[dc + e][row] = vv[e];
    }
    __syncthreads();
    v8f sc0 = {}, sc1 = {};
    #pragma unroll
    for (int kc = 0; kc < 4; ++kc) {  // B-frag: lane->key, elems->dh
      v16bf bk0 = ld_frag(&lds_k[lmod][kc * 32 + kbase],
                          &lds_k[lmod][kc * 32 + kbase + 16]);
      v16bf bk1 = ld_frag(&lds_k[16 + lmod][kc * 32 + kbase],
                          &lds_k[16 + lmod][kc * 32 + kbase + 16]);
      sc0 = __builtin_amdgcn_wmma_f32_16x16x32_bf16(
          false, qf[kc], false, bk0, (short)0, sc0, false, false);
      sc1 = __builtin_amdgcn_wmma_f32_16x16x32_bf16(
          false, qf[kc], false, bk1, (short)0, sc1, false, false);
    }
    // Online softmax; row m = r + 8*lhalf lives in one 16-lane half
    #pragma unroll
    for (int r = 0; r < 8; ++r) {
      float s0 = sc0[r] * scale, s1 = sc1[r] * scale;
      float mx = fmaxf(s0, s1);
      mx = fmaxf(mx, __shfl_xor(mx, 1, 32));
      mx = fmaxf(mx, __shfl_xor(mx, 2, 32));
      mx = fmaxf(mx, __shfl_xor(mx, 4, 32));
      mx = fmaxf(mx, __shfl_xor(mx, 8, 32));
      float nm   = fmaxf(mrow[r], mx);
      float corr = __expf(mrow[r] - nm);
      float p0 = __expf(s0 - nm), p1 = __expf(s1 - nm);
      float sum = p0 + p1;
      sum += __shfl_xor(sum, 1, 32);
      sum += __shfl_xor(sum, 2, 32);
      sum += __shfl_xor(sum, 4, 32);
      sum += __shfl_xor(sum, 8, 32);
      lrow[r] = lrow[r] * corr + sum;
      mrow[r] = nm;
      #pragma unroll
      for (int nt = 0; nt < 8; ++nt) o[nt][r] *= corr;
      int ml = r + lhalf * 8;
      lds_p[wave][ml][lmod]      = f2bf(p0);
      lds_p[wave][ml][16 + lmod] = f2bf(p1);
    }
    // PV: A-frag = P (lane->qrow, elems->key); B-frag = V (lane->dh)
    v16bf pf = ld_frag(&lds_p[wave][lmod][kbase],
                       &lds_p[wave][lmod][kbase + 16]);
    #pragma unroll
    for (int nt = 0; nt < 8; ++nt) {
      v16bf vf = ld_frag(&lds_vt[nt * 16 + lmod][kbase],
                         &lds_vt[nt * 16 + lmod][kbase + 16]);
      o[nt] = __builtin_amdgcn_wmma_f32_16x16x32_bf16(
          false, pf, false, vf, (short)0, o[nt], false, false);
    }
  }
  #pragma unroll
  for (int nt = 0; nt < 8; ++nt) {
    #pragma unroll
    for (int r = 0; r < 8; ++r) {
      int ml = r + lhalf * 8;
      size_t idx =
          (size_t)(b * Sc + q0 + ml) * Dc + h * DHc + nt * 16 + lmod;
      o_out[idx] = f2bf(o[nt][r] / lrow[r]);
    }
  }
}

// ---------------------------------------------------------------------------
// RMSNorm: one block per token row.
// ---------------------------------------------------------------------------
template <bool OUT_BF16>
__global__ __launch_bounds__(BDIM) void rmsnorm_kernel(
    const float* __restrict__ x, unsigned short* __restrict__ outB,
    float* __restrict__ outF)
{
  __shared__ float red[8];
  int row = blockIdx.x, tid = threadIdx.x;
  const float* xr = x + (size_t)row * Dc;
  float ss = 0.0f;
  for (int d = tid; d < Dc; d += BDIM) { float v2 = xr[d]; ss += v2 * v2; }
  ss += __shfl_xor(ss, 1, 32);
  ss += __shfl_xor(ss, 2, 32);
  ss += __shfl_xor(ss, 4, 32);
  ss += __shfl_xor(ss, 8, 32);
  ss += __shfl_xor(ss, 16, 32);
  if ((tid & 31) == 0) red[tid >> 5] = ss;
  __syncthreads();
  if (tid == 0) {
    float t = 0.0f;
    for (int i = 0; i < 8; ++i) t += red[i];
    red[0] = t;
  }
  __syncthreads();
  float sc = rsqrtf(red[0] / (float)Dc + 1e-6f);
  for (int d = tid; d < Dc; d += BDIM) {
    float v2 = xr[d] * sc;
    if constexpr (OUT_BF16) outB[(size_t)row * Dc + d] = f2bf(v2);
    else                    outF[(size_t)row * Dc + d] = v2;
  }
}

// ---------------------------------------------------------------------------
// Interleaved RoPE, in-place on bf16 q/k [B*S, D]. One pair / thread.
// ---------------------------------------------------------------------------
__global__ void rope_kernel(unsigned short* __restrict__ buf)
{
  int idx = blockIdx.x * blockDim.x + threadIdx.x;
  const int PH = Dc / 2;
  if (idx >= Bc * Sc * PH) return;
  int token = idx / PH, pc = idx % PH;
  int hh = pc / (DHc / 2), t = pc % (DHc / 2);
  int pos = token % Sc;
  // inv_freq = theta^(-2t/DH) = exp(-(2t/DH)*ln(10000))
  float ang = (float)pos *
      __expf((float)t * (-2.0f / (float)DHc) * 9.210340371976184f);
  float cs = cosf(ang), sn = sinf(ang);
  size_t base = (size_t)token * Dc + hh * DHc + 2 * t;
  float x1 = bf2f(buf[base]), x2 = bf2f(buf[base + 1]);
  buf[base]     = f2bf(x1 * cs - x2 * sn);
  buf[base + 1] = f2bf(x1 * sn + x2 * cs);
}

// ---------------------------------------------------------------------------
// Register replacement: stable order of valid tokens + register fill.
// ---------------------------------------------------------------------------
__global__ void scan_kernel(const float* __restrict__ mask,
                            int* __restrict__ perm, int* __restrict__ nvalid)
{
  int b = blockIdx.x;
  if (threadIdx.x == 0) {
    int n = 0;
    for (int s = 0; s < Sc; ++s)
      if (mask[b * Sc + s] >= -9000.0f) perm[b * Sc + n++] = s;
    nvalid[b] = n;
  }
}

__global__ __launch_bounds__(BDIM) void replace_kernel(
    const float* __restrict__ hs, const float* __restrict__ mask,
    const float* __restrict__ regs, const int* __restrict__ perm,
    const int* __restrict__ nvalid, float* __restrict__ x)
{
  int b = blockIdx.x / Sc, i = blockIdx.x % Sc;
  bool flip = mask[b * Sc + (Sc - 1 - i)] >= -9000.0f;  // reversed valid
  int n = nvalid[b];
  const float* src = nullptr;
  if (flip) {
    if (i < n) src = hs + (size_t)(b * Sc + perm[b * Sc + i]) * Dc;
  } else {
    src = regs + (size_t)(i % Rc) * Dc;
  }
  float* dst = x + (size_t)(b * Sc + i) * Dc;
  for (int d = threadIdx.x; d < Dc; d += BDIM)
    dst[d] = src ? src[d] : 0.0f;
}

__global__ void zero_kernel(float* p, int n) {
  int i = blockIdx.x * blockDim.x + threadIdx.x;
  if (i < n) p[i] = 0.0f;
}

// ---------------------------------------------------------------------------
extern "C" void kernel_launch(void* const* d_in, const int* in_sizes, int n_in,
                              void* d_out, int out_size, void* d_ws,
                              size_t ws_size, hipStream_t stream)
{
  (void)in_sizes; (void)n_in; (void)out_size; (void)ws_size;
  const float* hs   = (const float*)d_in[0];
  const float* mask = (const float*)d_in[1];
  const float* regs = (const float*)d_in[2];
  const float* Wq   = (const float*)d_in[3];
  const float* Wk   = (const float*)d_in[4];
  const float* Wv   = (const float*)d_in[5];
  const float* Wo   = (const float*)d_in[6];
  const float* W1   = (const float*)d_in[7];
  const float* W2   = (const float*)d_in[8];
  float* out = (float*)d_out;

  const size_t BS  = (size_t)Bc * Sc;
  const size_t BSD = BS * Dc;
  char* ws = (char*)d_ws;
  size_t off = 0;
  auto take = [&](size_t bytes) {
    char* p = ws + off;
    off = (off + bytes + 255) & ~(size_t)255;
    return p;
  };
  float*          x    = (float*)take(BSD * 4);
  unsigned short* nbuf = (unsigned short*)take(BSD * 2);
  unsigned short* qb   = (unsigned short*)take(BSD * 2);
  unsigned short* kb   = (unsigned short*)take(BSD * 2);
  unsigned short* vb   = (unsigned short*)take(BSD * 2);
  unsigned short* ab   = (unsigned short*)take(BSD * 2);
  unsigned short* h1   = (unsigned short*)take(BS * DFFc * 2);
  int*            perm = (int*)take(BS * 4);
  int*            nvld = (int*)take(Bc * 4);

  scan_kernel<<<Bc, 32, 0, stream>>>(mask, perm, nvld);
  replace_kernel<<<(int)BS, BDIM, 0, stream>>>(hs, mask, regs, perm, nvld, x);
  zero_kernel<<<((int)BS + 255) / 256, 256, 0, stream>>>(out + BSD, (int)BS);

  const int M = (int)BS;                       // 2048
  dim3 gD(Dc / 128, M / 64);                   // (30, 32)
  dim3 gF1(DFFc / 128, M / 64);                // (120, 32)
  const int ropeBlocks = (int)((BS * (Dc / 2) + BDIM - 1) / BDIM);

  for (int l = 0; l < 2; ++l) {
    const float* wq = Wq + (size_t)l * Dc * Dc;
    const float* wk = Wk + (size_t)l * Dc * Dc;
    const float* wv = Wv + (size_t)l * Dc * Dc;
    const float* wo = Wo + (size_t)l * Dc * Dc;
    const float* w1 = W1 + (size_t)l * Dc * DFFc;
    const float* w2 = W2 + (size_t)l * DFFc * Dc;

    rmsnorm_kernel<true><<<M, BDIM, 0, stream>>>(x, nbuf, nullptr);
    gemm_wmma<EPI_BF16><<<gD, BDIM, 0, stream>>>(nbuf, wq, qb, nullptr, nullptr, M, Dc, Dc);
    gemm_wmma<EPI_BF16><<<gD, BDIM, 0, stream>>>(nbuf, wk, kb, nullptr, nullptr, M, Dc, Dc);
    gemm_wmma<EPI_BF16><<<gD, BDIM, 0, stream>>>(nbuf, wv, vb, nullptr, nullptr, M, Dc, Dc);
    rope_kernel<<<ropeBlocks, BDIM, 0, stream>>>(qb);
    rope_kernel<<<ropeBlocks, BDIM, 0, stream>>>(kb);
    attn_kernel<<<dim3(Bc * Hc, Sc / 128), BDIM, 0, stream>>>(qb, kb, vb, ab);
    gemm_wmma<EPI_RESID><<<gD, BDIM, 0, stream>>>(ab, wo, nullptr, x, x, M, Dc, Dc);
    rmsnorm_kernel<true><<<M, BDIM, 0, stream>>>(x, nbuf, nullptr);
    gemm_wmma<EPI_GELU><<<gF1, BDIM, 0, stream>>>(nbuf, w1, h1, nullptr, nullptr, M, DFFc, Dc);
    gemm_wmma<EPI_RESID><<<gD, BDIM, 0, stream>>>(h1, w2, nullptr, x, x, M, Dc, DFFc);
  }
  rmsnorm_kernel<false><<<M, BDIM, 0, stream>>>(x, nullptr, out);
}
